// RecurrentModel_76459007803410
// MI455X (gfx1250) — matrix operand order repaired and checked by reading, not compiled, not run on hardware
//
#include <hip/hip_runtime.h>

typedef __attribute__((ext_vector_type(16))) __bf16 v16bf;
typedef __attribute__((ext_vector_type(8)))  __bf16 v8bf;
typedef __attribute__((ext_vector_type(8)))  float  v8f;

#define RNN_B 128
#define RNN_S 1000
#define RNN_I 64
#define RNN_H 512
#define REC_BLOCKS 32

static __device__ __forceinline__ v16bf concat8(v8bf lo, v8bf hi) {
  return __builtin_shufflevector(lo, hi, 0, 1, 2, 3, 4, 5, 6, 7,
                                         8, 9, 10, 11, 12, 13, 14, 15);
}

// ---------------------------------------------------------------------------
// Kernel A: convert W_hh and W_ih -> bf16, zero h0, reset barrier counter.
// ---------------------------------------------------------------------------
__global__ void __launch_bounds__(256) prep_kernel(
    const float* __restrict__ WhhF, const float* __restrict__ WihF,
    __bf16* __restrict__ WhhBf, __bf16* __restrict__ WihBf,
    __bf16* __restrict__ h0, unsigned* __restrict__ cnt) {
  const unsigned tid = blockIdx.x * 256u + threadIdx.x;
  if (tid < (unsigned)(RNN_H * RNN_H)) WhhBf[tid] = (__bf16)WhhF[tid];
  if (tid < (unsigned)(RNN_H * RNN_I)) WihBf[tid] = (__bf16)WihF[tid];
  if (tid < (unsigned)(RNN_B * RNN_H)) h0[tid] = (__bf16)0.0f;
  if (tid == 0) *cnt = 0u;
}

// ---------------------------------------------------------------------------
// Kernel B: x_proj = x @ W_ih^T + b_ih  (M=128000, N=512, K=64), one wave per
// 16x16 tile, two v_wmma_f32_16x16x32_bf16 per tile. Writes f32 into the
// core_output region of d_out (later overwritten in-place by h).
// A block's 8 waves share one M-tile, so the x tile stays hot in the WGP$.
// ---------------------------------------------------------------------------
__global__ void __launch_bounds__(256) xproj_kernel(
    const float* __restrict__ x, const __bf16* __restrict__ WihBf,
    const float* __restrict__ bih, float* __restrict__ core) {
  const int wave = threadIdx.x >> 5;
  const int lane = threadIdx.x & 31;
  const int sel  = lane >> 4;   // lane half
  const int l16  = lane & 15;
  const int id   = blockIdx.x * 8 + wave;   // tile id 0..255999
  const int mt   = id >> 5;                 // 0..7999
  const int nt   = id & 31;                 // 0..31
  const int arow = mt * 16 + l16;           // x row (b*S+s flattened)
  const int ncol = nt * 16 + l16;           // output column / W_ih row

  const float*  xr = x     + (size_t)arow * RNN_I;
  const __bf16* wr = WihBf + (size_t)ncol * RNN_I;

  v8f acc;
  const float bias = bih[ncol];
#pragma unroll
  for (int r = 0; r < 8; ++r) acc[r] = bias;

#pragma unroll
  for (int kk = 0; kk < 2; ++kk) {
    const int k0 = kk * 32;
    v16bf a;
    const float* p = xr + k0 + sel * 16;     // K {0..7} or {16..23}
    const float* q = p + 8;                  // K {8..15} or {24..31}
#pragma unroll
    for (int i = 0; i < 8; ++i) {
      a[i]     = (__bf16)p[i];
      a[8 + i] = (__bf16)q[i];
    }
    const v16bf b = *(const v16bf*)(wr + k0 + sel * 16);  // 16 contiguous K
    acc = __builtin_amdgcn_wmma_f32_16x16x32_bf16(
        false, a, false, b, (short)0, acc, false, false);
  }

#pragma unroll
  for (int r = 0; r < 8; ++r) {
    const int row = mt * 16 + r + sel * 8;
    core[(size_t)row * RNN_H + ncol] = acc[r];
  }
}

// ---------------------------------------------------------------------------
// Kernel C: persistent recurrence. 32 blocks x 8 waves = 256 waves, one wave
// per 16x16 tile of h (8 M-tiles x 32 N-tiles). W_hh fragments register-
// resident across all 1000 steps; h ping-pongs through L2 (bf16); two
// independent WMMA accumulation chains per step; prefetch of next step's
// x_proj; one global software barrier per step.
// ---------------------------------------------------------------------------
__global__ void __launch_bounds__(256) recurrent_kernel(
    const float* __restrict__ bhh, const __bf16* __restrict__ Whh,
    __bf16* __restrict__ hbuf0, __bf16* __restrict__ hbuf1,
    float* __restrict__ core, unsigned* __restrict__ cnt) {
  const int wave = threadIdx.x >> 5;
  const int lane = threadIdx.x & 31;
  const int sel  = lane >> 4;
  const int l16  = lane & 15;
  const int w    = blockIdx.x * 8 + wave;  // 0..255
  const int mt   = w >> 5;                 // 0..7
  const int nt   = w & 31;                 // 0..31
  const int arow = mt * 16 + l16;          // h row (batch index)
  const int ncol = nt * 16 + l16;          // output col / W_hh row

  // Preload B-matrix (W_hh) fragments: whole K=512 for this N-tile.
  v16bf bfrag[16];
#pragma unroll
  for (int ks = 0; ks < 16; ++ks)
    bfrag[ks] = *(const v16bf*)(Whh + (size_t)ncol * RNN_H + ks * 32 + sel * 16);

  const float bias = bhh[ncol];

  for (int t = 0; t < RNN_S; ++t) {
    const __bf16* hOld = (t & 1) ? hbuf1 : hbuf0;
    __bf16*       hNew = (t & 1) ? hbuf0 : hbuf1;

    // acc0 = x_proj[:, t, :] + b_hh ; acc1 = 0  (two independent chains)
    v8f acc0, acc1;
#pragma unroll
    for (int r = 0; r < 8; ++r) {
      const int m = mt * 16 + r + sel * 8;
      acc0[r] = core[((size_t)m * RNN_S + t) * RNN_H + ncol] + bias;
      acc1[r] = 0.0f;
    }

    // acc += h_old @ W_hh^T : 16 WMMAs over K=512, split into 2 chains.
#pragma unroll
    for (int ks = 0; ks < 16; ks += 2) {
      const __bf16* ap0 = hOld + (size_t)arow * RNN_H + ks * 32 + sel * 16;
      const __bf16* ap1 = ap0 + 32;
      v8bf c0 = *(const v8bf*)ap0;
      v8bf c1 = *(const v8bf*)(ap0 + 8);
      v8bf d0 = *(const v8bf*)ap1;
      v8bf d1 = *(const v8bf*)(ap1 + 8);
      acc0 = __builtin_amdgcn_wmma_f32_16x16x32_bf16(
          false, concat8(c0, c1), false, bfrag[ks], (short)0, acc0, false,
          false);
      acc1 = __builtin_amdgcn_wmma_f32_16x16x32_bf16(
          false, concat8(d0, d1), false, bfrag[ks + 1], (short)0, acc1, false,
          false);
    }

    // Prefetch next step's x_proj slots while results drain.
    if (t + 1 < RNN_S) {
#pragma unroll
      for (int r = 0; r < 8; ++r) {
        const int m = mt * 16 + r + sel * 8;
        __builtin_prefetch(core + ((size_t)m * RNN_S + (t + 1)) * RNN_H + ncol,
                           0, 1);
      }
    }

    // h = tanh(acc0 + acc1): f32 to core_output (overwrites x_proj slot read
    // by this same lane), bf16 to the next h buffer.
#pragma unroll
    for (int r = 0; r < 8; ++r) {
      const int m = mt * 16 + r + sel * 8;
      const float s = acc0[r] + acc1[r];
      const float e = __expf(2.0f * s);
      const float h = 1.0f - 2.0f / (e + 1.0f);   // tanh, saturates cleanly
      core[((size_t)m * RNN_S + t) * RNN_H + ncol] = h;
      hNew[(size_t)m * RNN_H + ncol] = (__bf16)h;
    }

    // Grid-wide barrier (monotonic counter; reset each launch by prep_kernel).
    __syncthreads();
    if (threadIdx.x == 0) {
      __builtin_amdgcn_fence(__ATOMIC_RELEASE, "agent");
      atomicAdd(cnt, 1u);
      const unsigned target = (unsigned)(t + 1) * (unsigned)REC_BLOCKS;
      while (__hip_atomic_load(cnt, __ATOMIC_RELAXED, __HIP_MEMORY_SCOPE_AGENT) <
             target)
        __builtin_amdgcn_s_sleep(1);
    }
    __syncthreads();
    __builtin_amdgcn_fence(__ATOMIC_ACQUIRE, "agent");  // invalidate stale L0
  }
}

// ---------------------------------------------------------------------------
// Kernel D: readouts. One wave per (b,s) row: two 512-length dots with b128
// loads, wave32 shuffle reduction, sigmoid on the second.
// ---------------------------------------------------------------------------
__global__ void __launch_bounds__(256) readout_kernel(
    const float* __restrict__ core, const float* __restrict__ Wro,
    const float* __restrict__ bro, const float* __restrict__ Wsig,
    const float* __restrict__ bsig, float* __restrict__ ro,
    float* __restrict__ sg) {
  const int row  = blockIdx.x * 8 + (threadIdx.x >> 5);  // 0..127999
  const int lane = threadIdx.x & 31;
  const float4* h4 = (const float4*)(core + (size_t)row * RNN_H);
  const float4* w0 = (const float4*)Wro;
  const float4* w1 = (const float4*)Wsig;
  float a0 = 0.0f, a1 = 0.0f;
#pragma unroll
  for (int i = 0; i < 4; ++i) {
    const int idx = i * 32 + lane;   // 128 float4 = 512 floats
    const float4 v = h4[idx];
    const float4 p = w0[idx];
    const float4 q = w1[idx];
    a0 += v.x * p.x + v.y * p.y + v.z * p.z + v.w * p.w;
    a1 += v.x * q.x + v.y * q.y + v.z * q.z + v.w * q.w;
  }
#pragma unroll
  for (int off = 16; off > 0; off >>= 1) {
    a0 += __shfl_xor(a0, off, 32);
    a1 += __shfl_xor(a1, off, 32);
  }
  if (lane == 0) {
    ro[row] = a0 + bro[0];
    const float z = a1 + bsig[0];
    sg[row] = 1.0f / (1.0f + __expf(-z));
  }
}

// ---------------------------------------------------------------------------
extern "C" void kernel_launch(void* const* d_in, const int* in_sizes, int n_in,
                              void* d_out, int out_size, void* d_ws,
                              size_t ws_size, hipStream_t stream) {
  const float* x    = (const float*)d_in[0];
  const float* Wih  = (const float*)d_in[1];
  const float* Whh  = (const float*)d_in[2];
  const float* bih  = (const float*)d_in[3];
  const float* bhh  = (const float*)d_in[4];
  const float* Wro  = (const float*)d_in[5];
  const float* bro  = (const float*)d_in[6];
  const float* Wsig = (const float*)d_in[7];
  const float* bsig = (const float*)d_in[8];

  float* out  = (float*)d_out;
  float* core = out;                                      // (B,S,H)
  float* ro   = out + (size_t)RNN_B * RNN_S * RNN_H;      // (B,S,1)
  float* sg   = ro + (size_t)RNN_B * RNN_S;               // (B,S,1)

  char* ws = (char*)d_ws;
  unsigned* cnt = (unsigned*)ws;
  size_t off = 256;
  __bf16* WhhBf = (__bf16*)(ws + off); off += (size_t)RNN_H * RNN_H * 2;
  __bf16* h0    = (__bf16*)(ws + off); off += (size_t)RNN_B * RNN_H * 2;
  __bf16* h1    = (__bf16*)(ws + off); off += (size_t)RNN_B * RNN_H * 2;
  __bf16* WihBf = (__bf16*)(ws + off);

  prep_kernel<<<1024, 256, 0, stream>>>(Whh, Wih, WhhBf, WihBf, h0, cnt);
  xproj_kernel<<<32000, 256, 0, stream>>>(x, WihBf, bih, core);
  recurrent_kernel<<<REC_BLOCKS, 256, 0, stream>>>(bhh, WhhBf, h0, h1, core,
                                                   cnt);
  readout_kernel<<<16000, 256, 0, stream>>>(core, Wro, bro, Wsig, bsig, ro, sg);
}
